// Net_76450417868816
// MI455X (gfx1250) — matrix operand order repaired and checked by reading, not compiled, not run on hardware
//
#include <hip/hip_runtime.h>

// Problem dimensions (from reference)
#define KD 784      // in_features
#define ND 1024     // out_features
#define MD 128      // batch
#define TITER 500   // noise iterations
#define KPAD 800    // K padded to multiple of 32 (25 k-steps)
#define LDSK 808    // LDS row stride in elements (x2B = 1616B, 16B aligned, conflict-skewed)
#define NTILE 32    // N columns per workgroup

typedef __attribute__((ext_vector_type(16))) __bf16 bf16x16;
typedef __attribute__((ext_vector_type(8)))  float  f32x8;
typedef __attribute__((ext_vector_type(4)))  float  f32x4;

union FragU {
  uint4   q[2];
  bf16x16 v;
};

__device__ __forceinline__ unsigned short f2bf(float f) {
  // round-to-nearest-even f32 -> bf16
  unsigned int u = __float_as_uint(f);
  u += 0x7FFFu + ((u >> 16) & 1u);
  return (unsigned short)(u >> 16);
}

// ---- Kernel 1: rnorm[o] = 1/||W[:,o]|| -------------------------------------
__global__ void k_colnorm(const float* __restrict__ W, float* __restrict__ rnorm) {
  int o = blockIdx.x * blockDim.x + threadIdx.x;
  if (o >= ND) return;
  float s = 0.f;
  for (int i = 0; i < KD; ++i) {
    float w = W[(size_t)i * ND + o];   // coalesced across o
    s += w * w;
  }
  rnorm[o] = rsqrtf(s);
}

// ---- Kernel 2: x -> bf16, K-padded to 800 ----------------------------------
__global__ void k_xbf(const float* __restrict__ x, unsigned short* __restrict__ xbf) {
  int idx = blockIdx.x * blockDim.x + threadIdx.x;
  if (idx >= MD * KPAD) return;
  int b = idx / KPAD, k = idx - b * KPAD;
  xbf[idx] = (k < KD) ? f2bf(x[b * KD + k]) : (unsigned short)0;
}

// ---- Kernel 3: rdenom[b,o] = 1/sqrt(sum_i x^2 * (W*rnorm)^2) ---------------
__global__ void k_rdenom(const float* __restrict__ x, const float* __restrict__ W,
                         const float* __restrict__ rnorm, float* __restrict__ rdenom) {
  int idx = blockIdx.x * blockDim.x + threadIdx.x;  // b*ND + o
  if (idx >= MD * ND) return;
  int b = idx >> 10, o = idx & (ND - 1);
  float acc = 0.f;
  for (int i = 0; i < KD; ++i) {
    float xv = x[b * KD + i];
    float wv = W[(size_t)i * ND + o];   // coalesced across o
    acc += (xv * xv) * (wv * wv);
  }
  float rn = rnorm[o];
  rdenom[idx] = rsqrtf(acc * rn * rn);
}

// ---- Kernel 4: main WMMA GEMM per (iteration, n-tile) ----------------------
// out[iter*128 + m, o] = (x @ (W_norm * (noise_iter - 1)))[m,o] * rdenom[m,o]
// Per-wave tiling: 2 m-tiles x 2 n-strips -> 4 WMMAs per k-step fed by
// 2 A-fragment global load pairs (unique per wave, no cross-wave duplication)
// and 2 B-fragment LDS load pairs.
__global__ void __launch_bounds__(128)
k_wmma_main(const float* __restrict__ noise, const float* __restrict__ W,
            const float* __restrict__ rnorm, const unsigned short* __restrict__ xbf,
            const float* __restrict__ rdenom, float* __restrict__ out) {
  __shared__ unsigned short Blds[NTILE * LDSK];   // 32 x 808 bf16 = 51.7 KB

  const int t     = threadIdx.x;   // 0..127 (4 waves)
  const int iter  = blockIdx.y;
  const int ncol0 = blockIdx.x * NTILE;

  // ---- Stage: Blds[n][k] = bf16((noise[iter][k][n0+n]-1) * W[k][n0+n] * rnorm[n0+n])
  {
    const int n4   = (t & 7) * 4;   // 0..28
    const int krow = t >> 3;        // 0..15
    f32x4 rn4 = *(const f32x4*)(rnorm + ncol0 + n4);
    const f32x4* np = (const f32x4*)(noise + ((size_t)iter * KD + krow) * ND + ncol0 + n4);
    const f32x4* wp = (const f32x4*)(W + (size_t)krow * ND + ncol0 + n4);
    const int stride4 = 16 * ND / 4;   // 16 rows in f32x4 units
    for (int k0 = 0; k0 < KD; k0 += 16) {
      int k = k0 + krow;
      f32x4 nv = __builtin_nontemporal_load(np);  // noise is stream-once: TH=NT
      f32x4 wv = *wp;                              // W is L2-hot
      if (k0 + 16 < KD) __builtin_prefetch((const void*)(np + stride4), 0, 0);
      Blds[(n4 + 0) * LDSK + k] = f2bf((nv.x - 1.f) * wv.x * rn4.x);
      Blds[(n4 + 1) * LDSK + k] = f2bf((nv.y - 1.f) * wv.y * rn4.y);
      Blds[(n4 + 2) * LDSK + k] = f2bf((nv.z - 1.f) * wv.z * rn4.z);
      Blds[(n4 + 3) * LDSK + k] = f2bf((nv.w - 1.f) * wv.w * rn4.w);
      np += stride4;
      wp += stride4;
    }
    // zero K padding 784..799
    for (int e = t; e < NTILE * 16; e += 128) {
      int n = e >> 4, k = KD + (e & 15);
      Blds[n * LDSK + k] = 0;
    }
  }
  __syncthreads();

  // ---- Compute: wave w -> m-tiles {2w, 2w+1}, n-strips {0,16}
  const int lane  = t & 31;
  const int w     = t >> 5;
  const int mbase = w * 2;            // m-tile index base (covers all 8 across 4 waves)

  f32x8 acc[2][2] = {};               // [mt][nf], 4 accumulators of 8 VGPRs

  // B fragment (32x16 bf16): lane = N col, half-lanes hold K [h*8,h*8+8) and +16
  const int bn = lane & 15;
  const int kB = (lane >> 4) * 8;
  // A fragment (16x32 bf16): lane = M row, half-lanes hold K [h*8,h*8+8) and +16
  const int m  = lane & 15;
  const int kA = (lane >> 4) * 8;

  const unsigned short* arow0 = xbf + (size_t)((mbase + 0) * 16 + m) * KPAD + kA;
  const unsigned short* arow1 = xbf + (size_t)((mbase + 1) * 16 + m) * KPAD + kA;
  const unsigned short* brow0 = &Blds[(0 + bn) * LDSK + kB];
  const unsigned short* brow1 = &Blds[(16 + bn) * LDSK + kB];

  for (int kk = 0; kk < KPAD; kk += 32) {
    FragU b0, b1, a0, a1;
    b0.q[0] = *(const uint4*)(brow0 + kk);        // K kk+kB    .. +7
    b0.q[1] = *(const uint4*)(brow0 + kk + 16);   // K kk+kB+16 .. +23
    b1.q[0] = *(const uint4*)(brow1 + kk);
    b1.q[1] = *(const uint4*)(brow1 + kk + 16);
    a0.q[0] = *(const uint4*)(arow0 + kk);        // K kk+kA    .. +7
    a0.q[1] = *(const uint4*)(arow0 + kk + 16);   // K kk+kA+16 .. +23
    a1.q[0] = *(const uint4*)(arow1 + kk);
    a1.q[1] = *(const uint4*)(arow1 + kk + 16);
    acc[0][0] = __builtin_amdgcn_wmma_f32_16x16x32_bf16(
        false, a0.v, false, b0.v, (short)0, acc[0][0], false, false);
    acc[0][1] = __builtin_amdgcn_wmma_f32_16x16x32_bf16(
        false, a0.v, false, b1.v, (short)0, acc[0][1], false, false);
    acc[1][0] = __builtin_amdgcn_wmma_f32_16x16x32_bf16(
        false, a1.v, false, b0.v, (short)0, acc[1][0], false, false);
    acc[1][1] = __builtin_amdgcn_wmma_f32_16x16x32_bf16(
        false, a1.v, false, b1.v, (short)0, acc[1][1], false, false);
  }

  // ---- Epilogue: scale by rdenom, streamed non-temporal stores
  const int nl = lane & 15;
  const int mh = lane >> 4;   // 0: rows r, 1: rows r+8
#pragma unroll
  for (int mt = 0; mt < 2; ++mt) {
#pragma unroll
    for (int nf = 0; nf < 2; ++nf) {
      const int col = ncol0 + nf * 16 + nl;
#pragma unroll
      for (int r = 0; r < 8; ++r) {
        int row = (mbase + mt) * 16 + mh * 8 + r;
        float v = acc[mt][nf][r] * rdenom[row * ND + col];
        __builtin_nontemporal_store(v, &out[((size_t)iter * MD + row) * ND + col]);
      }
    }
  }
}

extern "C" void kernel_launch(void* const* d_in, const int* in_sizes, int n_in,
                              void* d_out, int out_size, void* d_ws, size_t ws_size,
                              hipStream_t stream) {
  const float* x     = (const float*)d_in[0];   // [128,784]
  const float* W     = (const float*)d_in[1];   // [784,1024]
  const float* noise = (const float*)d_in[2];   // [500,784,1024]
  float* out = (float*)d_out;                   // [500*128,1024]

  char* ws = (char*)d_ws;
  float*          rnorm  = (float*)(ws);                    // 1024 f32   (4 KB)
  unsigned short* xbf    = (unsigned short*)(ws + 4096);    // 128*800 u16 (200 KB)
  float*          rdenom = (float*)(ws + 4096 + 204800);    // 128*1024 f32 (512 KB)

  k_colnorm<<<dim3((ND + 255) / 256), dim3(256), 0, stream>>>(W, rnorm);
  k_xbf<<<dim3((MD * KPAD + 255) / 256), dim3(256), 0, stream>>>(x, xbf);
  k_rdenom<<<dim3((MD * ND + 255) / 256), dim3(256), 0, stream>>>(x, W, rnorm, rdenom);
  k_wmma_main<<<dim3(ND / NTILE, TITER), dim3(128), 0, stream>>>(
      noise, W, rnorm, xbf, rdenom, out);
}
